// Sequence_32667521254179
// MI455X (gfx1250) — matrix operand order repaired and checked by reading, not compiled, not run on hardware
//
#include <hip/hip_runtime.h>
#include <hip/hip_bf16.h>

// ---------------------------------------------------------------------------
// 2-layer LSTM scan (B=1024, D=128, T=256, H=51) as a persistent bf16-WMMA
// kernel for gfx1250.  One wave owns a 16-row batch tile for all 256 steps;
// all weights live in LDS as bf16 [N][K]; gates use v_wmma_f32_16x16x32_bf16.
// x(t+1) is staged to LDS with global_load_async_to_lds_b32 (ASYNCcnt) while
// step t computes; outputs are streamed with non-temporal stores.
// ---------------------------------------------------------------------------

typedef __bf16 bf16_t;
typedef bf16_t v16bf __attribute__((ext_vector_type(16)));
typedef bf16_t v8bf  __attribute__((ext_vector_type(8)));
typedef float  v8f   __attribute__((ext_vector_type(8)));

#define BATCH 1024
#define DIN   128
#define TLEN  256
#define HID   51
#define GP    208   // 4*51 padded to 13 WMMA n-tiles
#define K1    192   // [x(128) | h1(pad 64)]
#define K2    128   // [h1(pad 64) | h2(pad 64)]
#define KL    64    // h2 padded

// LDS layout (bytes)
#define OFF_W1 0
#define SZ_W1  (GP * K1 * 2)          // 79872
#define OFF_W2 (OFF_W1 + SZ_W1)       // 79872
#define SZ_W2  (GP * K2 * 2)          // 53248
#define OFF_WL (OFF_W2 + SZ_W2)       // 133120
#define SZ_WL  (DIN * KL * 2)         // 16384
#define OFF_B1 (OFF_WL + SZ_WL)       // 149504
#define OFF_B2 (OFF_B1 + GP * 4)      // 150336
#define OFF_BL (OFF_B2 + GP * 4)      // 151168
#define OFF_PW (OFF_BL + DIN * 4)     // 151680
// per-wave region
#define PW_XH   0                     // bf16 [16][192]  x|h1 A-buffer
#define PW_H2C  6144                  // bf16 [16][128]  h1|h2 A-buffer
#define PW_C1   10240                 // f32  [16][64]
#define PW_C2   14336                 // f32  [16][64]
#define PW_GU   18432                 // f32  [16][208]  gates / prev-out union
#define PW_XSTG 31744                 // f32  [16][128]  async x staging
#define PW_SZ   39936
#define SMEM_TOTAL (OFF_PW + 4 * PW_SZ)  // 311424 < 320KB WGP LDS

__device__ __forceinline__ float sigf(float x)       { return 1.0f / (1.0f + __expf(-x)); }
__device__ __forceinline__ float tanhf_fast(float x) { return 2.0f / (1.0f + __expf(-2.0f * x)) - 1.0f; }

// Async copy one f32 from global to LDS; tracked by ASYNCcnt.
__device__ __forceinline__ void async_ld_f32(const float* gptr, float* lptr) {
  unsigned int laddr = (unsigned int)(unsigned long long)lptr;  // low 32 bits = LDS byte offset
  asm volatile("global_load_async_to_lds_b32 %0, %1, off"
               :: "v"(laddr), "v"(gptr)
               : "memory");
}
__device__ __forceinline__ void wait_asynccnt0() {
  asm volatile("s_wait_asynccnt 0x0" ::: "memory");
}

// A fragment: 16x32 bf16, lane L holds row L%16; half=L/16 picks K run
// half*16..half*16+15 (contiguous) within the 32-wide k-tile.
__device__ __forceinline__ v16bf load_a(const bf16_t* base, int row, int kstride,
                                        int kbase, int half) {
  union { v16bf v; v8bf h[2]; } u;
  const bf16_t* p = base + row * kstride + kbase + half * 16;
  u.h[0] = *(const v8bf*)(p);
  u.h[1] = *(const v8bf*)(p + 8);
  return u.v;
}

// B fragment: 32x16 bf16 from LDS weights stored [N][K]; lane L supplies
// column N = L%16 (+ntile*16), K run = ktile*32 + half*16 .. +15 (contiguous).
__device__ __forceinline__ v16bf load_b(const bf16_t* w, int col, int kstride, int kbase) {
  return *(const v16bf*)(w + col * kstride + kbase);
}

__device__ __forceinline__ v8f wmma_bf16(v16bf a, v16bf b, v8f c) {
  return __builtin_amdgcn_wmma_f32_16x16x32_bf16(false, a, false, b, (short)0, c, false, false);
}

__device__ __forceinline__ v8f vzero8() {
  v8f z;
#pragma unroll
  for (int v = 0; v < 8; ++v) z[v] = 0.0f;
  return z;
}

__global__ __launch_bounds__(128, 1)
void lstm_scan_wmma(const float* __restrict__ data_in,
                    const float* __restrict__ w_ih1, const float* __restrict__ w_hh1,
                    const float* __restrict__ b_ih1, const float* __restrict__ b_hh1,
                    const float* __restrict__ w_ih2, const float* __restrict__ w_hh2,
                    const float* __restrict__ b_ih2, const float* __restrict__ b_hh2,
                    const float* __restrict__ w_lin, const float* __restrict__ b_lin,
                    const float* __restrict__ w_out, const float* __restrict__ b_out,
                    float* __restrict__ out) {
  extern __shared__ char smem[];
  const int tid   = threadIdx.x;
  const int lane  = tid & 31;
  const int wave  = tid >> 5;
  const int col16 = lane & 15;
  const int half  = lane >> 4;

  bf16_t* W1  = (bf16_t*)(smem + OFF_W1);
  bf16_t* W2  = (bf16_t*)(smem + OFF_W2);
  bf16_t* WL  = (bf16_t*)(smem + OFF_WL);
  float*  B1s = (float*)(smem + OFF_B1);
  float*  B2s = (float*)(smem + OFF_B2);
  float*  BLs = (float*)(smem + OFF_BL);

  // per-wave persistent state
  char*   pw   = smem + OFF_PW + wave * PW_SZ;
  bf16_t* XH   = (bf16_t*)(pw + PW_XH);    // [16][192] x | h1
  bf16_t* H2C  = (bf16_t*)(pw + PW_H2C);   // [16][128] h1 | h2
  float*  C1s  = (float*)(pw + PW_C1);     // [16][64]
  float*  C2s  = (float*)(pw + PW_C2);     // [16][64]
  float*  GU   = (float*)(pw + PW_GU);     // [16][208] gates / prev-out
  float*  XSTG = (float*)(pw + PW_XSTG);   // [16][128] raw x staging

  const int bBase = blockIdx.x * 64 + wave * 16;

  // ---- kick off async staging of x(t=0) immediately ----
  for (int i = lane; i < 16 * DIN; i += 32) {
    int r = i >> 7, d = i & 127;
    async_ld_f32(&data_in[(size_t)(bBase + r) * (DIN * TLEN) + (size_t)d * TLEN], &XSTG[i]);
  }

  // ---- stage weights to LDS (bf16, [N][K], zero padded) ----
  for (int i = tid; i < GP * K1; i += 128) {
    int n = i / K1, k = i - n * K1;
    float v = 0.0f;
    if (n < 4 * HID) {
      if (k < DIN)               v = w_ih1[n * DIN + k];
      else if ((k - DIN) < HID)  v = w_hh1[n * HID + (k - DIN)];
    }
    W1[i] = (bf16_t)v;
  }
  for (int i = tid; i < GP * K2; i += 128) {
    int n = i / K2, k = i - n * K2;
    float v = 0.0f;
    if (n < 4 * HID) {
      if (k < 64)      { if (k < HID)        v = w_ih2[n * HID + k]; }
      else             { int kk = k - 64; if (kk < HID) v = w_hh2[n * HID + kk]; }
    }
    W2[i] = (bf16_t)v;
  }
  for (int i = tid; i < DIN * KL; i += 128) {
    int n = i / KL, k = i - n * KL;
    WL[i] = (bf16_t)((k < HID) ? w_lin[n * HID + k] : 0.0f);
  }
  for (int i = tid; i < GP; i += 128) {
    B1s[i] = (i < 4 * HID) ? (b_ih1[i] + b_hh1[i]) : 0.0f;
    B2s[i] = (i < 4 * HID) ? (b_ih2[i] + b_hh2[i]) : 0.0f;
  }
  if (tid < DIN) BLs[tid] = b_lin[tid];

  // ---- zero-init per-wave recurrent state ----
  for (int i = lane; i < 16 * K1; i += 32) XH[i]  = (bf16_t)0.0f;
  for (int i = lane; i < 16 * K2; i += 32) H2C[i] = (bf16_t)0.0f;
  for (int i = lane; i < 16 * 64; i += 32) { C1s[i] = 0.0f; C2s[i] = 0.0f; }
  for (int i = lane; i < 16 * GP; i += 32) GU[i] = 0.0f;   // prev out = 0

  __syncthreads();

  for (int t = 0; t < TLEN; ++t) {
    // ---- 1) consume staged x_t, NaN-impute with previous prediction ----
    wait_asynccnt0();
    for (int i = lane; i < 16 * DIN; i += 32) {
      int r = i >> 7, d = i & 127;
      float x = XSTG[i];
      float prev = GU[r * GP + d];
      x = __builtin_isnan(x) ? prev : x;
      XH[r * K1 + d] = (bf16_t)x;
    }
    // ---- issue async staging of x(t+1); overlaps with this step's WMMAs ----
    if (t + 1 < TLEN) {
      for (int i = lane; i < 16 * DIN; i += 32) {
        int r = i >> 7, d = i & 127;
        async_ld_f32(&data_in[(size_t)(bBase + r) * (DIN * TLEN) + (size_t)d * TLEN + (t + 1)],
                     &XSTG[i]);
      }
    }
    if (t + 32 < TLEN)  // warm future L2 lines (t is the contiguous axis)
      __builtin_prefetch(&data_in[(size_t)(bBase + (lane >> 1)) * (DIN * TLEN) +
                                  (size_t)(lane & 1) * 64 * TLEN + t + 32], 0, 1);
    __syncthreads();

    // ---- 2) cell1 gates: [16,192] x [192,208] (bias folded in later) ----
    v8f acc[13];
#pragma unroll
    for (int nt = 0; nt < 13; ++nt) acc[nt] = vzero8();
#pragma unroll
    for (int kt = 0; kt < 6; ++kt) {
      v16bf a = load_a(XH, col16, K1, kt * 32, half);
#pragma unroll
      for (int nt = 0; nt < 13; ++nt) {
        v16bf b = load_b(W1, nt * 16 + col16, K1, kt * 32 + half * 16);
        acc[nt] = wmma_bf16(a, b, acc[nt]);
      }
    }
#pragma unroll
    for (int nt = 0; nt < 13; ++nt)
#pragma unroll
      for (int v = 0; v < 8; ++v)
        GU[(v + 8 * half) * GP + nt * 16 + col16] = acc[nt][v];
    __syncthreads();

    // ---- 3) cell1 activations (adds bias here) ----
    for (int idx = lane; idx < 16 * HID; idx += 32) {
      int r = idx / HID, j = idx - r * HID;
      const float* g = &GU[r * GP];
      float gi = g[j]           + B1s[j];
      float gf = g[HID + j]     + B1s[HID + j];
      float gg = g[2 * HID + j] + B1s[2 * HID + j];
      float go = g[3 * HID + j] + B1s[3 * HID + j];
      float c  = C1s[r * 64 + j];
      float cn = sigf(gf) * c + sigf(gi) * tanhf_fast(gg);
      float hn = sigf(go) * tanhf_fast(cn);
      C1s[r * 64 + j] = cn;
      bf16_t hb = (bf16_t)hn;
      XH[r * K1 + DIN + j] = hb;   // h1 for next step's cell1
      H2C[r * K2 + j]      = hb;   // h1 for this step's cell2
    }
    __syncthreads();

    // ---- 4) cell2 gates: [16,128] x [128,208] ----
#pragma unroll
    for (int nt = 0; nt < 13; ++nt) acc[nt] = vzero8();
#pragma unroll
    for (int kt = 0; kt < 4; ++kt) {
      v16bf a = load_a(H2C, col16, K2, kt * 32, half);
#pragma unroll
      for (int nt = 0; nt < 13; ++nt) {
        v16bf b = load_b(W2, nt * 16 + col16, K2, kt * 32 + half * 16);
        acc[nt] = wmma_bf16(a, b, acc[nt]);
      }
    }
#pragma unroll
    for (int nt = 0; nt < 13; ++nt)
#pragma unroll
      for (int v = 0; v < 8; ++v)
        GU[(v + 8 * half) * GP + nt * 16 + col16] = acc[nt][v];
    __syncthreads();

    // ---- 5) cell2 activations ----
    for (int idx = lane; idx < 16 * HID; idx += 32) {
      int r = idx / HID, j = idx - r * HID;
      const float* g = &GU[r * GP];
      float gi = g[j]           + B2s[j];
      float gf = g[HID + j]     + B2s[HID + j];
      float gg = g[2 * HID + j] + B2s[2 * HID + j];
      float go = g[3 * HID + j] + B2s[3 * HID + j];
      float c  = C2s[r * 64 + j];
      float cn = sigf(gf) * c + sigf(gi) * tanhf_fast(gg);
      float hn = sigf(go) * tanhf_fast(cn);
      C2s[r * 64 + j] = cn;
      H2C[r * K2 + 64 + j] = (bf16_t)hn;   // h2 feeds linear + next cell2
    }
    __syncthreads();

    // ---- 6) linear: out = h2 @ Wlin^T + b  ([16,64] x [64,128]) ----
    v8f lacc[8];
#pragma unroll
    for (int nt = 0; nt < 8; ++nt) lacc[nt] = vzero8();
#pragma unroll
    for (int kt = 0; kt < 2; ++kt) {
      v16bf a = load_a(H2C, col16, K2, 64 + kt * 32, half);
#pragma unroll
      for (int nt = 0; nt < 8; ++nt) {
        v16bf b = load_b(WL, nt * 16 + col16, KL, kt * 32 + half * 16);
        lacc[nt] = wmma_bf16(a, b, lacc[nt]);
      }
    }
    __syncthreads();   // gates scratch consumed; reuse as prev-out
#pragma unroll
    for (int nt = 0; nt < 8; ++nt)
#pragma unroll
      for (int v = 0; v < 8; ++v) {
        int r = v + 8 * half;
        int d = nt * 16 + col16;
        float val = lacc[nt][v] + BLs[d];
        GU[r * GP + d] = val;  // prev-out for next step's imputation
        __builtin_nontemporal_store(
            val, &out[(size_t)(bBase + r) * (DIN * TLEN) + (size_t)d * TLEN + t]);
      }
    __syncthreads();
  }

  // ---- lab = sigmoid(h2 @ w_out^T + b_out) ----
  if (lane < 16) {
    int r = lane;
    float s = b_out[0];
    for (int j = 0; j < HID; ++j)
      s += (float)H2C[r * K2 + 64 + j] * w_out[j];
    out[(size_t)BATCH * DIN * TLEN + (bBase + r)] = sigf(s);
  }
}

extern "C" void kernel_launch(void* const* d_in, const int* in_sizes, int n_in,
                              void* d_out, int out_size, void* d_ws, size_t ws_size,
                              hipStream_t stream) {
  const float* data_in = (const float*)d_in[0];
  // d_in[1] = future (int scalar, always 0 here)
  const float* w_ih1 = (const float*)d_in[2];
  const float* w_hh1 = (const float*)d_in[3];
  const float* b_ih1 = (const float*)d_in[4];
  const float* b_hh1 = (const float*)d_in[5];
  const float* w_ih2 = (const float*)d_in[6];
  const float* w_hh2 = (const float*)d_in[7];
  const float* b_ih2 = (const float*)d_in[8];
  const float* b_hh2 = (const float*)d_in[9];
  const float* w_lin = (const float*)d_in[10];
  const float* b_lin = (const float*)d_in[11];
  const float* w_out = (const float*)d_in[12];
  const float* b_out = (const float*)d_in[13];
  float* out = (float*)d_out;

  dim3 grid(BATCH / 64);   // 16 workgroups, 4 waves each, 16 batch rows/wave
  dim3 block(128);
  lstm_scan_wmma<<<grid, block, SMEM_TOTAL, stream>>>(
      data_in, w_ih1, w_hh1, b_ih1, b_hh1, w_ih2, w_hh2, b_ih2, b_hh2,
      w_lin, b_lin, w_out, b_out, out);
}
